// AttentiveQuantizer_81758997447025
// MI455X (gfx1250) — matrix operand order repaired and checked by reading, not compiled, not run on hardware
//
#include <hip/hip_runtime.h>
#include <hip/hip_bf16.h>

// ---------------------------------------------------------------------------
// AttentiveQuantizer for MI455X (gfx1250, wave32, WMMA bf16)
//
//   X[16384,256]   = NCHW->NHWC latent (bf16)
//   q              = X @ wq^T                  (WMMA gemm)
//   kmat           = codebook @ wk^T           (WMMA gemm)
//   v              = codebook @ wv^T           (WMMA gemm, f32 out + bf16 vT)
//   fused flash:     logit = q@kmat^T/16 (written once), online softmax,
//                    argmax, O = P@V, quantized = (hard - soft) + soft
// ---------------------------------------------------------------------------

typedef __attribute__((ext_vector_type(16))) __bf16 v16bf;
typedef __attribute__((ext_vector_type(8)))  __bf16 v8bf;
typedef __attribute__((ext_vector_type(8)))  float  v8f;

#define CIN 256
#define KCB 4096
#define NHW 16384
#define HW  1024

static __device__ __forceinline__ v8f wmma_acc(v16bf a, v16bf b, v8f c) {
  return __builtin_amdgcn_wmma_f32_16x16x32_bf16(
      /*neg_a=*/false, a, /*neg_b=*/false, b,
      /*c_mod=*/(short)0, c, /*reuse_a=*/false, /*reuse_b=*/false);
}

// A-matrix fragment (16x32 bf16, M x K): lane holds row (lane&15);
// lanes<16: K [0..15]; lanes>=16: K [16..31].
static __device__ __forceinline__ v16bf load_afrag(const __bf16* row, int c0, int lane) {
  const int ab = (lane & 16) ? 16 : 0;
  v8bf lo = *(const v8bf*)(row + c0 + ab);
  v8bf hi = *(const v8bf*)(row + c0 + ab + 8);
  return __builtin_shufflevector(lo, hi, 0,1,2,3,4,5,6,7,8,9,10,11,12,13,14,15);
}

// ---------------------------------------------------------------------------
// NCHW f32 -> [p=n*1024+h*32+w][c] bf16, LDS tile transpose for coalescing
// ---------------------------------------------------------------------------
__global__ __launch_bounds__(256)
void pack_latent(const float* __restrict__ latent, __bf16* __restrict__ X) {
  __shared__ float tile[32][33];
  const int p0 = blockIdx.x * 32;
  const int c0 = blockIdx.y * 32;
  const int tx = threadIdx.x, ty = threadIdx.y;
  const int n = p0 >> 10;
  const int hw0 = p0 & (HW - 1);
#pragma unroll
  for (int k = 0; k < 4; ++k) {
    const int c = c0 + ty + k * 8;
    tile[ty + k * 8][tx] = latent[((size_t)n * CIN + c) * HW + hw0 + tx];
  }
  __syncthreads();
#pragma unroll
  for (int k = 0; k < 4; ++k) {
    const int p = p0 + ty + k * 8;
    X[(size_t)p * CIN + c0 + tx] = (__bf16)tile[tx][ty + k * 8];
  }
}

__global__ __launch_bounds__(256)
void f32_to_bf16_k(const float* __restrict__ in, __bf16* __restrict__ out, int n) {
  const int i = blockIdx.x * blockDim.x + threadIdx.x;
  if (i < n) out[i] = (__bf16)in[i];
}

// ---------------------------------------------------------------------------
// C[m,n] = sum_c A[m,c]*B[n,c]   (A: MxK bf16 rowmajor, B: NxK bf16 rowmajor)
// One wave computes a 16(M) x 64(N) tile; block = 4 waves -> 64x64.
// Optional outputs: bf16 rowmajor, f32 rowmajor, bf16 transposed [N x M].
// ---------------------------------------------------------------------------
__global__ __launch_bounds__(128)
void gemm_bf16_tn(const __bf16* __restrict__ A, const __bf16* __restrict__ B,
                  int M, int N, int Kd,
                  __bf16* __restrict__ Cb, float* __restrict__ Cf,
                  __bf16* __restrict__ Ct) {
  const int lane = threadIdx.x & 31;
  const int wv = threadIdx.x >> 5;
  const int m0 = blockIdx.x * 64 + wv * 16;
  const int n0 = blockIdx.y * 64;
  const int colL = lane & 15;
  const int b16 = (lane & 16) ? 16 : 0;
  v8f acc[4] = {};
  const __bf16* arow = A + (size_t)(m0 + colL) * Kd;
  for (int c0 = 0; c0 < Kd; c0 += 32) {
    v16bf a = load_afrag(arow, c0, lane);
#pragma unroll
    for (int t = 0; t < 4; ++t) {
      v16bf b = *(const v16bf*)(B + (size_t)(n0 + t * 16 + colL) * Kd + c0 + b16);
      acc[t] = wmma_acc(a, b, acc[t]);
    }
  }
  const int off = (lane & 16) ? 8 : 0;
#pragma unroll
  for (int t = 0; t < 4; ++t)
#pragma unroll
    for (int r = 0; r < 8; ++r) {
      const int mm = m0 + r + off;
      const int nn = n0 + t * 16 + colL;
      const float val = acc[t][r];
      if (Cb) Cb[(size_t)mm * N + nn] = (__bf16)val;
      if (Cf) Cf[(size_t)mm * N + nn] = val;
      if (Ct) Ct[(size_t)nn * M + mm] = (__bf16)val;
    }
}

// ---------------------------------------------------------------------------
// Fused: logits write + online softmax + argmax + P@V + quantized/idx stores.
// One wave owns 16 rows; 8 waves per block; grid = NHW/128.
// ---------------------------------------------------------------------------
__global__ __launch_bounds__(256, 1)
void flash_quant(const __bf16* __restrict__ qb, const __bf16* __restrict__ km,
                 const __bf16* __restrict__ vT, const float* __restrict__ vF,
                 float* __restrict__ logit, float* __restrict__ idx_out,
                 float* __restrict__ quant) {
  __shared__ float sS[8 * 16 * 36];   // per-wave 16x32 score tile (stride 36)
  __shared__ float sRow[8 * 16];      // per-wave per-row scalar broadcast
  const int lane = threadIdx.x & 31;
  const int wv = threadIdx.x >> 5;
  const int rowbase = (blockIdx.x * 8 + wv) * 16;
  const int colL = lane & 15;
  const int hi = (lane & 16) ? 1 : 0;
  const int off = hi * 8;
  const int b16 = hi * 16;
  float* sw = sS + wv * (16 * 36);
  float* sr = sRow + wv * 16;

  // cache the 16x256 q tile as 8 A-frags (reused for all 4096 keys)
  v16bf aq[8];
  const __bf16* qrow = qb + (size_t)(rowbase + colL) * CIN;
#pragma unroll
  for (int c = 0; c < 8; ++c) aq[c] = load_afrag(qrow, c * 32, lane);

  v8f O[16] = {};                 // 16 rows x 256 cols accumulator (D-layout)
  float m_run = -3.0e38f;         // row = colL (duplicated across lane pair)
  float s_run = 0.f;
  float amv[8];
  int ami[8];
#pragma unroll
  for (int r = 0; r < 8; ++r) { amv[r] = -3.0e38f; ami[r] = 0; }

  for (int j = 0; j < KCB; j += 32) {
    // ---- S = q @ kmat^T for 32 keys ----
    v8f S0 = {}, S1 = {};
    const __bf16* k0 = km + (size_t)(j + colL) * CIN + b16;
    const __bf16* k1 = km + (size_t)(j + 16 + colL) * CIN + b16;
#pragma unroll
    for (int c = 0; c < 8; ++c) {
      v16bf b0 = *(const v16bf*)(k0 + c * 32);
      v16bf b1 = *(const v16bf*)(k1 + c * 32);
      S0 = wmma_acc(aq[c], b0, S0);
      S1 = wmma_acc(aq[c], b1, S1);
    }
#pragma unroll
    for (int r = 0; r < 8; ++r) { S0[r] *= 0.0625f; S1[r] *= 0.0625f; }

    // ---- raw logits out (written exactly once) ----
    float* lg = logit + (size_t)(rowbase + off) * KCB + j + colL;
#pragma unroll
    for (int r = 0; r < 8; ++r) {
      lg[(size_t)r * KCB] = S0[r];
      lg[(size_t)r * KCB + 16] = S1[r];
    }

    // ---- running per-lane argmax (strict > keeps lowest key per lane) ----
#pragma unroll
    for (int r = 0; r < 8; ++r) {
      if (S0[r] > amv[r]) { amv[r] = S0[r]; ami[r] = j + colL; }
      if (S1[r] > amv[r]) { amv[r] = S1[r]; ami[r] = j + 16 + colL; }
    }

    // ---- D-layout -> LDS -> A-layout (row-major per lane) ----
#pragma unroll
    for (int r = 0; r < 8; ++r) {
      sw[(r + off) * 36 + colL]      = S0[r];
      sw[(r + off) * 36 + 16 + colL] = S1[r];
    }
    asm volatile("s_wait_dscnt 0x0" ::: "memory");
    const float* rp = sw + colL * 36 + b16;   // row colL, keys b16..b16+15
    float pv[16];
    {
      float4 f0 = *(const float4*)(rp + 0);
      float4 f1 = *(const float4*)(rp + 4);
      float4 f2 = *(const float4*)(rp + 8);
      float4 f3 = *(const float4*)(rp + 12);
      pv[0]=f0.x; pv[1]=f0.y; pv[2]=f0.z; pv[3]=f0.w;
      pv[4]=f1.x; pv[5]=f1.y; pv[6]=f1.z; pv[7]=f1.w;
      pv[8]=f2.x; pv[9]=f2.y; pv[10]=f2.z; pv[11]=f2.w;
      pv[12]=f3.x; pv[13]=f3.y; pv[14]=f3.z; pv[15]=f3.w;
    }

    // ---- online softmax update (row = colL, lane pair combined by xor16) ----
    float lmax = pv[0];
#pragma unroll
    for (int i = 1; i < 16; ++i) lmax = fmaxf(lmax, pv[i]);
    const float cmax = fmaxf(lmax, __shfl_xor(lmax, 16, 32));
    const float m_new = fmaxf(m_run, cmax);
    const float fsc = __expf(m_run - m_new);
    float lsum = 0.f;
    v16bf P;
#pragma unroll
    for (int i = 0; i < 16; ++i) {
      const float e = __expf(pv[i] - m_new);
      lsum += e;
      P[i] = (__bf16)e;                       // pv order == A-frag element order
    }
    s_run = s_run * fsc + lsum + __shfl_xor(lsum, 16, 32);
    m_run = m_new;

    // broadcast per-row rescale factor into D-layout ordering
    if (!hi) sr[colL] = fsc;
    asm volatile("s_wait_dscnt 0x0" ::: "memory");
    float4 fa = *(const float4*)(sr + off);
    float4 fb = *(const float4*)(sr + off + 4);
    const float fv[8] = {fa.x, fa.y, fa.z, fa.w, fb.x, fb.y, fb.z, fb.w};
#pragma unroll
    for (int t = 0; t < 16; ++t)
#pragma unroll
      for (int r = 0; r < 8; ++r) O[t][r] *= fv[r];

    // ---- O += P @ V(chunk) ----
#pragma unroll
    for (int t = 0; t < 16; ++t) {
      v16bf b = *(const v16bf*)(vT + (size_t)(t * 16 + colL) * KCB + j + b16);
      O[t] = wmma_acc(P, b, O[t]);
    }

    if (j + 32 < KCB) {
      __builtin_prefetch(km + (size_t)(j + 32 + colL) * CIN, 0, 1);
      __builtin_prefetch(vT + (size_t)colL * KCB + j + 32, 0, 1);
    }
  }

  // ---- 1/sum per row in D-layout ----
  if (!hi) sr[colL] = s_run;
  asm volatile("s_wait_dscnt 0x0" ::: "memory");
  float4 sa = *(const float4*)(sr + off);
  float4 sb = *(const float4*)(sr + off + 4);
  const float svv[8] = {sa.x, sa.y, sa.z, sa.w, sb.x, sb.y, sb.z, sb.w};
  float rs[8];
#pragma unroll
  for (int r = 0; r < 8; ++r) rs[r] = 1.f / svv[r];

  // ---- cross-lane argmax within each 16-lane half (ties -> lowest index) ----
#pragma unroll
  for (int step = 1; step < 16; step <<= 1) {
#pragma unroll
    for (int r = 0; r < 8; ++r) {
      const float vo = __shfl_xor(amv[r], step, 32);
      const int   io = __shfl_xor(ami[r], step, 32);
      if (vo > amv[r] || (vo == amv[r] && io < ami[r])) { amv[r] = vo; ami[r] = io; }
    }
  }
  if (colL == 0) {
#pragma unroll
    for (int r = 0; r < 8; ++r)
      idx_out[rowbase + r + off] = (float)(ami[r] & 255);   // uint8 cast
  }

  // ---- quantized = (hard_v - soft_v) + soft_v, scattered to NCHW ----
#pragma unroll
  for (int r = 0; r < 8; ++r) {
    const int p = rowbase + r + off;
    const int n = p >> 10;
    const int hw = p & (HW - 1);
    const float* vrow = vF + (size_t)ami[r] * CIN;
    float* qo = quant + (size_t)n * CIN * HW + hw;
#pragma unroll
    for (int t = 0; t < 16; ++t) {
      const int c = t * 16 + colL;
      const float soft = O[t][r] * rs[r];
      const float hard = vrow[c];
      qo[(size_t)c * HW] = (hard - soft) + soft;
    }
  }
}

// ---------------------------------------------------------------------------
extern "C" void kernel_launch(void* const* d_in, const int* in_sizes, int n_in,
                              void* d_out, int out_size, void* d_ws, size_t ws_size,
                              hipStream_t stream) {
  (void)in_sizes; (void)n_in; (void)out_size; (void)ws_size;
  const float* latent   = (const float*)d_in[0];
  // d_in[1] = temperature (unused in deterministic forward)
  const float* codebook = (const float*)d_in[2];
  const float* wq       = (const float*)d_in[3];
  const float* wk       = (const float*)d_in[4];
  const float* wv       = (const float*)d_in[5];

  float* out       = (float*)d_out;
  float* out_quant = out;                                   // 16*256*32*32
  float* out_idx   = out_quant + (size_t)16 * CIN * HW;     // 16384
  float* out_logit = out_idx + NHW;                         // 16384*4096
  float* out_v     = out_logit + (size_t)NHW * KCB;         // 4096*256

  char* w = (char*)d_ws;
  auto take = [&](size_t bytes) {
    char* p = w;
    w += (bytes + 255) & ~(size_t)255;
    return p;
  };
  __bf16* X   = (__bf16*)take((size_t)NHW * CIN * 2);
  __bf16* qbf = (__bf16*)take((size_t)NHW * CIN * 2);
  __bf16* cbb = (__bf16*)take((size_t)KCB * CIN * 2);
  __bf16* kmb = (__bf16*)take((size_t)KCB * CIN * 2);
  __bf16* vTb = (__bf16*)take((size_t)KCB * CIN * 2);   // [CIN][KCB]
  __bf16* wqb = (__bf16*)take((size_t)CIN * CIN * 2);
  __bf16* wkb = (__bf16*)take((size_t)CIN * CIN * 2);
  __bf16* wvb = (__bf16*)take((size_t)CIN * CIN * 2);

  pack_latent<<<dim3(NHW / 32, CIN / 32), dim3(32, 8), 0, stream>>>(latent, X);
  f32_to_bf16_k<<<(CIN * CIN + 255) / 256, 256, 0, stream>>>(wq, wqb, CIN * CIN);
  f32_to_bf16_k<<<(CIN * CIN + 255) / 256, 256, 0, stream>>>(wk, wkb, CIN * CIN);
  f32_to_bf16_k<<<(CIN * CIN + 255) / 256, 256, 0, stream>>>(wv, wvb, CIN * CIN);
  f32_to_bf16_k<<<(KCB * CIN + 255) / 256, 256, 0, stream>>>(codebook, cbb, KCB * CIN);

  // q = X @ wq^T                       [16384,256]
  gemm_bf16_tn<<<dim3(NHW / 64, CIN / 64), 128, 0, stream>>>(
      X, wqb, NHW, CIN, CIN, qbf, nullptr, nullptr);
  // kmat = codebook @ wk^T             [4096,256] bf16
  gemm_bf16_tn<<<dim3(KCB / 64, CIN / 64), 128, 0, stream>>>(
      cbb, wkb, KCB, CIN, CIN, kmb, nullptr, nullptr);
  // v = codebook @ wv^T                f32 -> output, bf16 transposed -> vT
  gemm_bf16_tn<<<dim3(KCB / 64, CIN / 64), 128, 0, stream>>>(
      cbb, wvb, KCB, CIN, CIN, nullptr, out_v, vTb);

  flash_quant<<<NHW / 128, 256, 0, stream>>>(
      qbf, kmb, vTb, out_v, out_logit, out_idx, out_quant);
}
